// MSARowAttentionWithPairBias_81338090651654
// MI455X (gfx1250) — matrix operand order repaired and checked by reading, not compiled, not run on hardware
//
#include <hip/hip_runtime.h>
#include <hip/hip_bf16.h>

// MSARowAttentionWithPairBias for MI455X (gfx1250, wave32, WMMA bf16).
// Workspace requirement: ~86.7 MB (5 x 16MB bf16 activations + 2MB bias + 0.65MB weights).

#define S_DIM 128
#define R_DIM 256
#define CM_DIM 256
#define CZ_DIM 128
#define C_DIM 32
#define H_DIM 8
#define EPS 1e-5f
#define QK_SCALE 0.17677669529663687f  // 1/sqrt(32)

typedef __attribute__((ext_vector_type(16))) __bf16 v16bf;
typedef __attribute__((ext_vector_type(8)))  __bf16 bf16x8;
typedef __attribute__((ext_vector_type(8)))  float  v8f;

// ---------------------------------------------------------------------------
// Kernel 1: convert + transpose weights to bf16.  wT[w][n][k] = W[w][k][n]
// w = 0:Wq 1:Wk 2:Wv 3:Wg 4:Wo   (all [256][256] f32, k-major)
// ---------------------------------------------------------------------------
__global__ void prep_weights_kernel(const float* __restrict__ Wq,
                                    const float* __restrict__ Wk,
                                    const float* __restrict__ Wv,
                                    const float* __restrict__ Wg,
                                    const float* __restrict__ Wo,
                                    __bf16* __restrict__ wT) {
    int b = blockIdx.x;
    int widx = b >> 8;
    int n = b & 255;
    int k = threadIdx.x;
    const float* W = (widx == 0) ? Wq : (widx == 1) ? Wk : (widx == 2) ? Wv
                   : (widx == 3) ? Wg : Wo;
    wT[((size_t)widx << 16) + (size_t)n * 256 + k] = (__bf16)W[(size_t)k * 256 + n];
}

// ---------------------------------------------------------------------------
// Kernel 2: LayerNorm over CM for m -> bf16 mn.  One 256-thread block per row.
// ---------------------------------------------------------------------------
__global__ void ln_m_kernel(const float* __restrict__ m,
                            const float* __restrict__ w,
                            const float* __restrict__ b,
                            __bf16* __restrict__ mn) {
    int row = blockIdx.x;
    int t = threadIdx.x;
    int lane = t & 31, wave = t >> 5;
    float x = m[(size_t)row * CM_DIM + t];
    float s = x, sq = x * x;
#pragma unroll
    for (int o = 16; o > 0; o >>= 1) {
        s  += __shfl_xor(s,  o, 32);
        sq += __shfl_xor(sq, o, 32);
    }
    __shared__ float red[16];
    if (lane == 0) { red[wave] = s; red[8 + wave] = sq; }
    __syncthreads();
    float tot = 0.f, totq = 0.f;
#pragma unroll
    for (int i = 0; i < 8; ++i) { tot += red[i]; totq += red[8 + i]; }
    float mu   = tot * (1.0f / CM_DIM);
    float var  = totq * (1.0f / CM_DIM) - mu * mu;
    float rinv = rsqrtf(var + EPS);
    mn[(size_t)row * CM_DIM + t] = (__bf16)((x - mu) * rinv * w[t] + b[t]);
}

// ---------------------------------------------------------------------------
// Kernel 3: LayerNorm(z) and pair bias = zn @ Wz -> bias[h][q][k] (f32).
// One wave (32 lanes) per (q,k) pair; 4 z-elements per lane.
// ---------------------------------------------------------------------------
__global__ void zbias_kernel(const float* __restrict__ z,
                             const float* __restrict__ lnw,
                             const float* __restrict__ lnb,
                             const float* __restrict__ Wz,
                             float* __restrict__ bias) {
    int tid  = threadIdx.x;
    int lane = tid & 31, wave = tid >> 5;
    size_t pair = (size_t)blockIdx.x * 8 + wave;           // q*256 + k
    const float* zr = z + pair * CZ_DIM;
    float xv[4];
#pragma unroll
    for (int j = 0; j < 4; ++j) xv[j] = zr[lane * 4 + j];
    float s = xv[0] + xv[1] + xv[2] + xv[3];
#pragma unroll
    for (int o = 16; o > 0; o >>= 1) s += __shfl_xor(s, o, 32);
    float mu = s * (1.0f / CZ_DIM);
    float ss = 0.f;
#pragma unroll
    for (int j = 0; j < 4; ++j) { float d = xv[j] - mu; ss += d * d; }
#pragma unroll
    for (int o = 16; o > 0; o >>= 1) ss += __shfl_xor(ss, o, 32);
    float rinv = rsqrtf(ss * (1.0f / CZ_DIM) + EPS);
    float zn[4];
#pragma unroll
    for (int j = 0; j < 4; ++j)
        zn[j] = (xv[j] - mu) * rinv * lnw[lane * 4 + j] + lnb[lane * 4 + j];
    float acc[H_DIM];
#pragma unroll
    for (int h = 0; h < H_DIM; ++h) acc[h] = 0.f;
#pragma unroll
    for (int j = 0; j < 4; ++j) {
        const float* wr = Wz + (size_t)(lane * 4 + j) * H_DIM;
#pragma unroll
        for (int h = 0; h < H_DIM; ++h) acc[h] += zn[j] * wr[h];
    }
#pragma unroll
    for (int h = 0; h < H_DIM; ++h) {
#pragma unroll
        for (int o = 16; o > 0; o >>= 1) acc[h] += __shfl_xor(acc[h], o, 32);
    }
    if (lane == 0) {
#pragma unroll
        for (int h = 0; h < H_DIM; ++h)
            bias[((size_t)h << 16) + pair] = acc[h];
    }
}

// ---------------------------------------------------------------------------
// Kernel 4: FUSED Q/K/V/G projection GEMM.
// A = mn [32768 x 256] bf16 row-major; wT = 4 transposed weights [w][n][k].
// One wave per 16x16 output tile position; each A-fragment feeds 4 WMMAs
// (one per weight matrix) -> 32 v_wmma per wave, A traffic amortized 4x.
// Epilogues: Q scaled -> [S][H][R][C]; K -> [S][H][R][C]; V -> [S][H][C][R];
// G = sigmoid(x+bg) -> [row][col].
// ---------------------------------------------------------------------------
__global__ void proj_qkvg_kernel(const __bf16* __restrict__ A,
                                 const __bf16* __restrict__ wT,
                                 __bf16* __restrict__ qo,
                                 __bf16* __restrict__ ko,
                                 __bf16* __restrict__ vto,
                                 __bf16* __restrict__ go_,
                                 const float* __restrict__ bg) {
    int tid  = threadIdx.x;
    int lane = tid & 31, wave = tid >> 5;
    int tile = blockIdx.x * 8 + wave;          // 32768 tiles total
    int mt = tile >> 4;                        // 0..2047
    int nt = tile & 15;                        // 0..15
    int r16 = lane & 15, hi = lane >> 4;

    const __bf16* arow = A + (size_t)(mt * 16 + r16) * 256;
    const __bf16* brow = wT + (size_t)(nt * 16 + r16) * 256;   // + w*65536
    int aoff = hi * 8;     // A: e0-7 -> K=8*hi+e ; e8-15 -> K=16+8*hi+(e-8)
    int boff = hi * 8;     // B: e0-7 -> K=8*hi+e ; e8-15 -> K=16+8*hi+(e-8)

    v8f acc[4];
#pragma unroll
    for (int w = 0; w < 4; ++w) acc[w] = (v8f){};

#pragma unroll
    for (int ch = 0; ch < 8; ++ch) {
        v16bf a;
        bf16x8* ap = reinterpret_cast<bf16x8*>(&a);
        ap[0] = *reinterpret_cast<const bf16x8*>(arow + ch * 32 + aoff);
        ap[1] = *reinterpret_cast<const bf16x8*>(arow + ch * 32 + aoff + 16);
#pragma unroll
        for (int w = 0; w < 4; ++w) {
            v16bf bb;
            bf16x8* bp = reinterpret_cast<bf16x8*>(&bb);
            const __bf16* bw = brow + (size_t)w * 65536 + ch * 32 + boff;
            bp[0] = *reinterpret_cast<const bf16x8*>(bw);
            bp[1] = *reinterpret_cast<const bf16x8*>(bw + 16);
            acc[w] = __builtin_amdgcn_wmma_f32_16x16x32_bf16(
                false, a, false, bb, (short)0, acc[w], false, false);
        }
    }

    int col = nt * 16 + r16;                   // 0..255
    int h = col >> 5, c = col & 31;
#pragma unroll
    for (int r = 0; r < 8; ++r) {
        int grow = mt * 16 + r + hi * 8;       // 0..32767
        int s = grow >> 8, rq = grow & 255;
        size_t shr = ((size_t)s * H_DIM + h);
        qo [(shr * R_DIM + rq) * C_DIM + c] = (__bf16)(acc[0][r] * QK_SCALE);
        ko [(shr * R_DIM + rq) * C_DIM + c] = (__bf16)acc[1][r];
        vto[(shr * C_DIM + c) * R_DIM + rq] = (__bf16)acc[2][r];
        float gv = 1.0f / (1.0f + __expf(-(acc[3][r] + bg[col])));
        go_[(size_t)grow * 256 + col] = (__bf16)gv;
    }
}

// ---------------------------------------------------------------------------
// Kernel 5: attention core.  Block = (s, h, 64-query tile), 256 threads.
// Dynamic LDS: 64x256 f32 logits (64KB) + 64x256 bf16 probs (32KB) = 96KB.
// Phase 1: logits = Q K^T + bias (WMMA).  Phase 2: softmax.
// Phase 3: O = P V (WMMA), gate with G, store g*o bf16 in [S][R][H*C].
// ---------------------------------------------------------------------------
__global__ void attn_kernel(const __bf16* __restrict__ q,
                            const __bf16* __restrict__ k,
                            const __bf16* __restrict__ vt,
                            const __bf16* __restrict__ g,
                            const float* __restrict__ bias,
                            __bf16* __restrict__ go) {
    extern __shared__ __align__(16) char smem[];
    float*  lg = reinterpret_cast<float*>(smem);                 // [64][256]
    __bf16* pr = reinterpret_cast<__bf16*>(smem + 64 * 256 * 4); // [64][256]

    int tid  = threadIdx.x;
    int lane = tid & 31, wave = tid >> 5;
    int blk = blockIdx.x;                 // s*32 + h*4 + qb
    int qb = blk & 3;
    int h  = (blk >> 2) & 7;
    int s  = blk >> 5;
    int qbase = qb * 64;
    int r16 = lane & 15, hi = lane >> 4;

    const __bf16* Qb = q  + ((size_t)s * H_DIM + h) * R_DIM * C_DIM;  // [r][c]
    const __bf16* Kb = k  + ((size_t)s * H_DIM + h) * R_DIM * C_DIM;  // [r][c]
    const __bf16* Vb = vt + ((size_t)s * H_DIM + h) * C_DIM * R_DIM;  // [c][r]

    // ---- Phase 1: logits (4 m-tiles x 16 n-tiles = 64 tiles; 8 per wave) ----
#pragma unroll
    for (int t = 0; t < 8; ++t) {
        int tileId = wave * 8 + t;
        int mt = tileId >> 4;             // 0..3
        int nt = tileId & 15;             // 0..15
        v16bf a, bb;
        bf16x8* ap = reinterpret_cast<bf16x8*>(&a);
        bf16x8* bp = reinterpret_cast<bf16x8*>(&bb);
        const __bf16* ar = Qb + (size_t)(qbase + mt * 16 + r16) * C_DIM;
        ap[0] = *reinterpret_cast<const bf16x8*>(ar + hi * 8);
        ap[1] = *reinterpret_cast<const bf16x8*>(ar + hi * 8 + 16);
        const __bf16* br = Kb + (size_t)(nt * 16 + r16) * C_DIM;
        bp[0] = *reinterpret_cast<const bf16x8*>(br + hi * 8);
        bp[1] = *reinterpret_cast<const bf16x8*>(br + hi * 8 + 16);
        v8f c = {};
        c = __builtin_amdgcn_wmma_f32_16x16x32_bf16(false, a, false, bb,
                                                    (short)0, c, false, false);
        const float* bh = bias + ((size_t)h * R_DIM + qbase + mt * 16) * R_DIM
                               + nt * 16 + r16;
#pragma unroll
        for (int r = 0; r < 8; ++r) {
            int rr = r + hi * 8;
            lg[(mt * 16 + rr) * 256 + nt * 16 + r16] = c[r] + bh[(size_t)rr * R_DIM];
        }
    }
    __syncthreads();

    // ---- Phase 2: row softmax (4 threads per row, 64 elements each) ----
    {
        int rloc = tid >> 2;
        int part = tid & 3;
        float* rowp = lg + rloc * 256 + part * 64;
        float mx = -3.0e38f;
#pragma unroll 8
        for (int i = 0; i < 64; ++i) mx = fmaxf(mx, rowp[i]);
        mx = fmaxf(mx, __shfl_xor(mx, 1, 32));
        mx = fmaxf(mx, __shfl_xor(mx, 2, 32));
        float sm = 0.f;
#pragma unroll 8
        for (int i = 0; i < 64; ++i) {
            float e = __expf(rowp[i] - mx);
            rowp[i] = e;
            sm += e;
        }
        sm += __shfl_xor(sm, 1, 32);
        sm += __shfl_xor(sm, 2, 32);
        float inv = 1.0f / sm;
        __bf16* pp = pr + rloc * 256 + part * 64;
#pragma unroll 8
        for (int i = 0; i < 64; ++i) pp[i] = (__bf16)(rowp[i] * inv);
    }
    __syncthreads();

    // ---- Phase 3: O = P V  (4 m-tiles x 2 n-tiles = 8 tiles; one per wave) ----
    {
        int mt = wave >> 1;               // 0..3
        int nt = wave & 1;                // 0..1
        v8f o = {};
#pragma unroll
        for (int ch = 0; ch < 8; ++ch) {
            v16bf a, bb;
            bf16x8* ap = reinterpret_cast<bf16x8*>(&a);
            bf16x8* bp = reinterpret_cast<bf16x8*>(&bb);
            const __bf16* ar = pr + (size_t)(mt * 16 + r16) * 256 + ch * 32;
            ap[0] = *reinterpret_cast<const bf16x8*>(ar + hi * 8);
            ap[1] = *reinterpret_cast<const bf16x8*>(ar + hi * 8 + 16);
            const __bf16* br = Vb + (size_t)(nt * 16 + r16) * R_DIM + ch * 32 + hi * 8;
            bp[0] = *reinterpret_cast<const bf16x8*>(br);
            bp[1] = *reinterpret_cast<const bf16x8*>(br + 16);
            o = __builtin_amdgcn_wmma_f32_16x16x32_bf16(false, a, false, bb,
                                                        (short)0, o, false, false);
        }
        int colc = nt * 16 + r16;         // c in 0..31
#pragma unroll
        for (int r = 0; r < 8; ++r) {
            int rq = qbase + mt * 16 + r + hi * 8;
            size_t gi = ((size_t)s * R_DIM + rq) * (H_DIM * C_DIM) + h * C_DIM + colc;
            float gv = (float)g[gi];
            go[gi] = (__bf16)(gv * o[r]);
        }
    }
}

// ---------------------------------------------------------------------------
// Kernel 6: output projection GEMM: out = go @ Wo + bo  (f32 out).
// ---------------------------------------------------------------------------
__global__ void gemm_out_kernel(const __bf16* __restrict__ A,
                                const __bf16* __restrict__ BT,
                                float* __restrict__ dst,
                                const float* __restrict__ bo) {
    int tid  = threadIdx.x;
    int lane = tid & 31, wave = tid >> 5;
    int tile = blockIdx.x * 8 + wave;
    int mt = tile >> 4;
    int nt = tile & 15;
    int r16 = lane & 15, hi = lane >> 4;

    const __bf16* arow = A  + (size_t)(mt * 16 + r16) * 256;
    const __bf16* brow = BT + (size_t)(nt * 16 + r16) * 256;
    int aoff = hi * 8;
    int boff = hi * 8;

    v8f acc = {};
#pragma unroll
    for (int ch = 0; ch < 8; ++ch) {
        v16bf a, bb;
        bf16x8* ap = reinterpret_cast<bf16x8*>(&a);
        bf16x8* bp = reinterpret_cast<bf16x8*>(&bb);
        ap[0] = *reinterpret_cast<const bf16x8*>(arow + ch * 32 + aoff);
        ap[1] = *reinterpret_cast<const bf16x8*>(arow + ch * 32 + aoff + 16);
        bp[0] = *reinterpret_cast<const bf16x8*>(brow + ch * 32 + boff);
        bp[1] = *reinterpret_cast<const bf16x8*>(brow + ch * 32 + boff + 16);
        acc = __builtin_amdgcn_wmma_f32_16x16x32_bf16(false, a, false, bb,
                                                      (short)0, acc, false, false);
    }

    int col = nt * 16 + r16;
    float bias = bo[col];
#pragma unroll
    for (int r = 0; r < 8; ++r) {
        int grow = mt * 16 + r + hi * 8;
        dst[(size_t)grow * 256 + col] = acc[r] + bias;
    }
}

// ---------------------------------------------------------------------------
// Host launcher
// ---------------------------------------------------------------------------
extern "C" void kernel_launch(void* const* d_in, const int* in_sizes, int n_in,
                              void* d_out, int out_size, void* d_ws, size_t ws_size,
                              hipStream_t stream) {
    (void)in_sizes; (void)n_in; (void)out_size; (void)ws_size;
    const float* m      = (const float*)d_in[0];
    const float* z      = (const float*)d_in[1];
    const float* ln_m_w = (const float*)d_in[2];
    const float* ln_m_b = (const float*)d_in[3];
    const float* ln_z_w = (const float*)d_in[4];
    const float* ln_z_b = (const float*)d_in[5];
    const float* Wz     = (const float*)d_in[6];
    const float* Wq     = (const float*)d_in[7];
    const float* Wk     = (const float*)d_in[8];
    const float* Wv     = (const float*)d_in[9];
    const float* Wg     = (const float*)d_in[10];
    const float* bg     = (const float*)d_in[11];
    const float* Wo     = (const float*)d_in[12];
    const float* bo     = (const float*)d_in[13];
    float* outp = (float*)d_out;

    // Workspace layout (bytes); mn is reused as go after projections finish.
    const size_t SRxCM = (size_t)S_DIM * R_DIM * CM_DIM;     // 8,388,608 elements
    char* ws = (char*)d_ws;
    size_t off = 0;
    __bf16* mn   = (__bf16*)(ws + off); off += SRxCM * 2;            // 16 MB (also "go")
    __bf16* wT   = (__bf16*)(ws + off); off += 5 * 65536 * 2;        // 0.64 MB
    __bf16* qb   = (__bf16*)(ws + off); off += SRxCM * 2;            // 16 MB
    __bf16* kb   = (__bf16*)(ws + off); off += SRxCM * 2;            // 16 MB
    __bf16* vtb  = (__bf16*)(ws + off); off += SRxCM * 2;            // 16 MB
    __bf16* gb   = (__bf16*)(ws + off); off += SRxCM * 2;            // 16 MB
    float*  bias = (float*)(ws + off);  off += (size_t)H_DIM * R_DIM * R_DIM * 4; // 2 MB
    __bf16* go   = mn;   // alias: mn fully consumed before attention writes go

    // 1. Weight prep (transpose + bf16)
    prep_weights_kernel<<<5 * 256, 256, 0, stream>>>(Wq, Wk, Wv, Wg, Wo, wT);
    // 2. LayerNorm m
    ln_m_kernel<<<S_DIM * R_DIM, 256, 0, stream>>>(m, ln_m_w, ln_m_b, mn);
    // 3. Pair bias
    zbias_kernel<<<(R_DIM * R_DIM) / 8, 256, 0, stream>>>(z, ln_z_w, ln_z_b, Wz, bias);
    // 4. Fused Q/K/V/G projection (single pass over mn, 4 WMMAs per A-fragment)
    proj_qkvg_kernel<<<4096, 256, 0, stream>>>(mn, wT, qb, kb, vtb, gb, bg);
    // 5. Attention (96 KB dynamic LDS)
    attn_kernel<<<S_DIM * H_DIM * 4, 256, 64 * 256 * 4 + 64 * 256 * 2, stream>>>(
        qb, kb, vtb, gb, bias, go);
    // 6. Output projection
    gemm_out_kernel<<<4096, 256, 0, stream>>>(go, wT + 4 * 65536, outp, bo);
}
